// DynamicHypergraphTower_32246614459126
// MI455X (gfx1250) — matrix operand order repaired and checked by reading, not compiled, not run on hardware
//
#include <hip/hip_runtime.h>
#include <math.h>
#include <stdint.h>

typedef __attribute__((ext_vector_type(16))) _Float16 v16h;
typedef __attribute__((ext_vector_type(8)))  float    v8f;
typedef __attribute__((ext_vector_type(4)))  unsigned int u32x4;
typedef __attribute__((ext_vector_type(4)))  int      i32x4;
typedef __attribute__((ext_vector_type(8)))  int      i32x8;

#define DEV __device__ __forceinline__

namespace cfg {
constexpr int B   = 4;
constexpr int S   = 32;
constexpr int NE  = 50;
constexpr int D   = 512;
constexpr int DSS = 32;    // state dim
constexpr int DC  = 4;     // conv width
constexpr int DI  = 1024;  // inner dim
constexpr int BN  = B * NE;        // 200
constexpr int BS  = B * S;         // 128
constexpr int TOK = BN * S;        // 6400
constexpr int XP  = 2 * DSS + 1;   // 65
}

DEV float sigmoidf_(float x) { return 1.f / (1.f + __expf(-x)); }
DEV float siluf_(float x)    { return x * sigmoidf_(x); }

DEV float block_reduce_sum(float v) {
  __shared__ float sm[32];
  const int lane = threadIdx.x & 31;
  const int w    = threadIdx.x >> 5;
  #pragma unroll
  for (int o = 16; o > 0; o >>= 1) v += __shfl_xor(v, o);
  if (lane == 0) sm[w] = v;
  __syncthreads();
  float r;
  const int nw = blockDim.x >> 5;
  if (w == 0) {
    r = (lane < nw) ? sm[lane] : 0.f;
    #pragma unroll
    for (int o = 16; o > 0; o >>= 1) r += __shfl_xor(r, o);
    if (lane == 0) sm[0] = r;
  }
  __syncthreads();
  r = sm[0];
  __syncthreads();
  return r;
}

// ---------------------------------------------------------------------------
// Tensor Data Mover: DMA a 16(n) x 32(k) f32 tile of W into LDS.
// D# per CDNA5 ISA: group0 = {count/lds_addr/global_addr/type},
// group1 = {data_size, tensor_dim0/1, tile_dim0/1/2, tensor_dim0_stride}.
// tensor_dim* set to the remaining extents so edge tiles zero-fill via TDM OOB.
// ---------------------------------------------------------------------------
DEV void tdm_load_tile16x32(const float* gsrc, unsigned lds_byte_off,
                            int k_rem, int n_rem, int pitch)
{
  const unsigned long long ga = (unsigned long long)(uintptr_t)gsrc;
  u32x4 g0;
  g0[0] = 1u;                                            // count=1, user mode
  g0[1] = lds_byte_off;                                  // lds_addr
  g0[2] = (unsigned)(ga & 0xffffffffull);                // global_addr[95:64]
  g0[3] = (unsigned)((ga >> 32) & 0x1ffffffull)          // global_addr[120:96]
        | (2u << 30);                                    // type = 2 ("image")
  const unsigned td0 = (unsigned)k_rem;                  // tensor_dim0 (elems)
  const unsigned td1 = (unsigned)n_rem;                  // tensor_dim1
  i32x8 g1;
  g1[0] = (int)(2u << 16);                               // wg_mask=0, data_size=4B
  g1[1] = (int)((td0 & 0xffffu) << 16);                  // tensor_dim0[15:0]
  g1[2] = (int)((td0 >> 16) | ((td1 & 0xffffu) << 16));  // dim0[31:16] | dim1[15:0]
  g1[3] = (int)((td1 >> 16) | (32u << 16));              // dim1[31:16] | tile_dim0=32
  g1[4] = 16;                                            // tile_dim1=16, tile_dim2=0
  g1[5] = pitch;                                         // tensor_dim0_stride[31:0]
  g1[6] = 0;                                             // stride hi | dim1_stride lo
  g1[7] = 0;
  const i32x4 z4 = {0, 0, 0, 0};
#if defined(__clang_major__) && (__clang_major__ >= 23)
  const i32x8 z8 = {0, 0, 0, 0, 0, 0, 0, 0};
  __builtin_amdgcn_tensor_load_to_lds(g0, g1, z4, z4, z8, 0);
#else
  __builtin_amdgcn_tensor_load_to_lds(g0, g1, z4, z4, 0);
#endif
}

// ---------------------------------------------------------------------------
// WMMA GEMM:  Y[M,N] = X[M,K] @ W[N,K]^T (+bias[N]) (+res[M,N]);  K % 32 == 0.
// Block = 4 waves covering 64 M-rows x 16 N-cols. The shared 16x32 W tile is
// staged by TDM (double buffered, s_wait_tensorcnt); A fragments are four
// unconditional b128 loads per lane (row index clamped; edge rows masked at
// the store).  One v_wmma_f32_16x16x32_f16 per K-chunk per wave.
// ---------------------------------------------------------------------------
__global__ void k_gemm_wmma(const float* __restrict__ X, const float* __restrict__ W,
                            const float* __restrict__ bias, const float* __restrict__ res,
                            float* __restrict__ Y, int M, int N, int K)
{
  __shared__ float tileW[2][16 * 32];      // 2 x 2KB, TDM destination (offset 0 / 2048)

  const int wave = threadIdx.x >> 5;
  const int lane = threadIdx.x & 31;
  const int grp  = lane >> 4;
  const int lm   = lane & 15;
  const int nt   = blockIdx.x;
  const int mt   = blockIdx.y * 4 + wave;
  const int arow = mt * 16 + lm;
  const int arow_c = (arow < M) ? arow : (M - 1);
  const float* xrow = X + (size_t)arow_c * K;
  const float* wtile = W + (size_t)nt * 16 * K;
  const int n_rem = N - nt * 16;
  const int nchunks = K >> 5;

  if (wave == 0)
    tdm_load_tile16x32(wtile, 0u, K, n_rem, K);

  v8f acc = {};
  for (int kc = 0; kc < nchunks; ++kc) {
    const int k0  = kc << 5;
    const int cur = kc & 1;
    if (wave == 0) {
      if (kc + 1 < nchunks) {
        tdm_load_tile16x32(wtile + (k0 + 32), (unsigned)((cur ^ 1) * 2048),
                           K - (k0 + 32), n_rem, K);
        __builtin_amdgcn_s_wait_tensorcnt((short)1);   // chunk kc complete
      } else {
        __builtin_amdgcn_s_wait_tensorcnt((short)0);
      }
    }
    __syncthreads();                                   // LDS tile visible to all waves

    __builtin_prefetch(xrow + k0 + 96, 0, 0);          // global_prefetch_b8 (A stream)

    // A fragment (ISA A layout): lane<16 holds K {0..7,16..23}, lane>=16 {8..15,24..31}
    const float4 a0 = *(const float4*)(xrow + k0 + 8 * grp);
    const float4 a1 = *(const float4*)(xrow + k0 + 8 * grp + 4);
    const float4 a2 = *(const float4*)(xrow + k0 + 16 + 8 * grp);
    const float4 a3 = *(const float4*)(xrow + k0 + 16 + 8 * grp + 4);
    v16h a, b;
    a[0] = (_Float16)a0.x;  a[1] = (_Float16)a0.y;
    a[2] = (_Float16)a0.z;  a[3] = (_Float16)a0.w;
    a[4] = (_Float16)a1.x;  a[5] = (_Float16)a1.y;
    a[6] = (_Float16)a1.z;  a[7] = (_Float16)a1.w;
    a[8] = (_Float16)a2.x;  a[9] = (_Float16)a2.y;
    a[10] = (_Float16)a2.z; a[11] = (_Float16)a2.w;
    a[12] = (_Float16)a3.x; a[13] = (_Float16)a3.y;
    a[14] = (_Float16)a3.z; a[15] = (_Float16)a3.w;

    // B fragment: lane holds column n=lane%16, K in same order as A
    #pragma unroll
    for (int i = 0; i < 16; ++i)
      b[i] = (_Float16)tileW[cur][lm * 32 + 8 * grp + i + (i & 8)];

    acc = __builtin_amdgcn_wmma_f32_16x16x32_f16(false, a, false, b, (short)0, acc,
                                                 false, false);
    __syncthreads();                                   // done reading tile[cur]
  }

  #pragma unroll
  for (int r = 0; r < 8; ++r) {
    const int m = mt * 16 + r + 8 * grp;   // C/D layout: VGPR r -> M = r + 8*(lane/16)
    const int n = nt * 16 + lm;            //             lane%16 -> N
    if (m < M && n < N) {
      float v = acc[r];
      if (bias) v += bias[n];
      if (res)  v += res[(size_t)m * N + n];
      Y[(size_t)m * N + n] = v;
    }
  }
}

// ---------------------------------------------------------------------------
// LayerNorm over last dim (one block per row; in-place safe)
// ---------------------------------------------------------------------------
__global__ void k_layernorm(const float* __restrict__ x, const float* __restrict__ g,
                            const float* __restrict__ b, float* __restrict__ y,
                            int rows, int Dm)
{
  const int row = blockIdx.x;
  if (row >= rows) return;
  const float* xr = x + (size_t)row * Dm;
  float s = 0.f;
  for (int i = threadIdx.x; i < Dm; i += blockDim.x) s += xr[i];
  const float mean = block_reduce_sum(s) / (float)Dm;
  float vs = 0.f;
  for (int i = threadIdx.x; i < Dm; i += blockDim.x) { const float d = xr[i] - mean; vs += d * d; }
  const float var = block_reduce_sum(vs) / (float)Dm;
  const float inv = rsqrtf(var + 1e-5f);
  float* yr = y + (size_t)row * Dm;
  for (int i = threadIdx.x; i < Dm; i += blockDim.x)
    yr[i] = (xr[i] - mean) * inv * g[i] + b[i];
}

// ---------------------------------------------------------------------------
// Pipeline elementwise / reduction kernels
// ---------------------------------------------------------------------------
__global__ void k_led(const float* __restrict__ inc, const float* __restrict__ scene,
                      float* __restrict__ led)
{
  using namespace cfg;
  const int i = blockIdx.x * blockDim.x + threadIdx.x;
  if (i >= BN * D) return;
  const int bn = i / D, d = i % D;
  const int b  = bn / NE;
  const float* ir = inc + (size_t)bn * S;
  float acc = 0.f, pr = 0.f;
  for (int s = 0; s < S; ++s) { const float r = ir[s]; pr += r; acc += r * scene[((size_t)b * S + s) * D + d]; }
  led[i] = acc / fmaxf(pr, 1e-8f);
}

__global__ void k_hinit_comb(const float* __restrict__ t1, const float* __restrict__ namep,
                             const float* __restrict__ type_emb, const int* __restrict__ entt,
                             const unsigned char* __restrict__ mask, float* __restrict__ h)
{
  using namespace cfg;
  const int i = blockIdx.x * blockDim.x + threadIdx.x;
  if (i >= BN * D) return;
  const int bn = i / D, d = i % D;
  const float m = mask[bn] ? 1.f : 0.f;
  h[i] = (t1[i] + type_emb[(size_t)entt[bn] * D + d] + namep[i]) * m;
}

__global__ void k_mean_ent(const float* __restrict__ inc, const unsigned char* __restrict__ mask,
                           const float* __restrict__ h, float* __restrict__ me)
{
  using namespace cfg;
  const int i = blockIdx.x * blockDim.x + threadIdx.x;
  if (i >= BS * D) return;
  const int bs = i / D, d = i % D;
  const int b = bs / S, s = bs % S;
  float acc = 0.f, ws = 0.f;
  for (int n = 0; n < NE; ++n) {
    const int bn = b * NE + n;
    const float r = inc[(size_t)bn * S + s] * (mask[bn] ? 1.f : 0.f);
    ws += r;
    acc += r * h[(size_t)bn * D + d];
  }
  me[i] = acc / fmaxf(ws, 1.f);
}

__global__ void k_eall_pre(const float* __restrict__ tA, const float* __restrict__ tB,
                           const float* __restrict__ he_emb, const int* __restrict__ edge,
                           float* __restrict__ ec)
{
  using namespace cfg;
  const int i = blockIdx.x * blockDim.x + threadIdx.x;
  if (i >= BS * D) return;
  const int bs = i / D, d = i % D;
  ec[i] = tA[i] + tB[i] + he_emb[(size_t)edge[bs] * D + d];
}

// out[b,i,j] = scale * dot(Q[b,i,:], Kk[b,j,:])   (float4 vectorized)
__global__ void k_dot(const float* __restrict__ Q, const float* __restrict__ Kk,
                      float* __restrict__ out, int nq, int nk, float scale)
{
  using namespace cfg;
  const int idx = blockIdx.x * blockDim.x + threadIdx.x;
  if (idx >= B * nq * nk) return;
  const int b = idx / (nq * nk);
  const int r = idx % (nq * nk);
  const int i = r / nk, j = r % nk;
  const float4* q = (const float4*)(Q  + (size_t)(b * nq + i) * D);
  const float4* k = (const float4*)(Kk + (size_t)(b * nk + j) * D);
  float acc = 0.f;
  for (int d = 0; d < D / 4; ++d) {
    const float4 a = q[d], c = k[d];
    acc += a.x * c.x + a.y * c.y + a.z * c.z + a.w * c.w;
  }
  out[idx] = acc * scale;
}

__global__ void k_mpre(const float* __restrict__ attn, const float* __restrict__ v,
                       float* __restrict__ out)
{
  using namespace cfg;
  const int i = blockIdx.x * blockDim.x + threadIdx.x;
  if (i >= BN * S * D) return;
  const int bns = i / D, d = i % D;
  const int b = bns / (NE * S);
  const int s = bns % S;
  out[i] = sigmoidf_(attn[bns]) * v[(size_t)(b * S + s) * D + d];
}

__global__ void k_arc_softmax(const float* __restrict__ arcS, const float* __restrict__ inc,
                              const unsigned char* __restrict__ mask,
                              float* __restrict__ attn_arc, float* __restrict__ anyv)
{
  using namespace cfg;
  const int idx = blockIdx.x * blockDim.x + threadIdx.x;
  if (idx >= BN * S) return;
  const int bn = idx / S, s = idx % S;
  const float mf = mask[bn] ? 1.f : 0.f;
  const float* sc = arcS + (size_t)bn * S;
  const bool ps = (inc[(size_t)bn * S + s] * mf) > 0.f;
  float e[S];
  float mx = -1e30f; bool any = false;
  for (int t = 0; t < S; ++t) {
    const bool valid = (t < s) && ps && ((inc[(size_t)bn * S + t] * mf) > 0.f);
    const float v = valid ? (sc[t] - 0.1f * (float)(s - t)) : -1e30f;
    e[t] = v;
    if (valid) { any = true; mx = fmaxf(mx, v); }
  }
  float sum = 0.f;
  for (int t = 0; t < S; ++t) {
    const float ex = any ? __expf(e[t] - mx) : 0.f;
    e[t] = ex; sum += ex;
  }
  const float invs = any ? (1.f / sum) : 0.f;
  for (int t = 0; t < S; ++t) attn_arc[(size_t)idx * S + t] = e[t] * invs;
  anyv[idx] = any ? 1.f : 0.f;
}

__global__ void k_arc_msg(const float* __restrict__ attn_arc, const float* __restrict__ pv,
                          float* __restrict__ out)
{
  using namespace cfg;
  const int i = blockIdx.x * blockDim.x + threadIdx.x;
  if (i >= BN * S * D) return;
  const int bns = i / D, d = i % D;
  const int b = bns / (NE * S);
  const float* w = attn_arc + (size_t)bns * S;
  float acc = 0.f;
  for (int t = 0; t < S; ++t) acc += w[t] * pv[(size_t)(b * S + t) * D + d];
  out[i] = acc;
}

__global__ void k_gelu(float* __restrict__ x, int n)
{
  const int i = blockIdx.x * blockDim.x + threadIdx.x;
  if (i >= n) return;
  const float v = x[i];
  x[i] = 0.5f * v * (1.f + erff(v * 0.70710678118654752f));
}

__global__ void k_softmax4(float* __restrict__ w, int rows)
{
  const int r = blockIdx.x * blockDim.x + threadIdx.x;
  if (r >= rows) return;
  float v0 = w[r * 4 + 0], v1 = w[r * 4 + 1], v2 = w[r * 4 + 2], v3 = w[r * 4 + 3];
  const float mx = fmaxf(fmaxf(v0, v1), fmaxf(v2, v3));
  v0 = __expf(v0 - mx); v1 = __expf(v1 - mx); v2 = __expf(v2 - mx); v3 = __expf(v3 - mx);
  const float inv = 1.f / (v0 + v1 + v2 + v3);
  w[r * 4 + 0] = v0 * inv; w[r * 4 + 1] = v1 * inv; w[r * 4 + 2] = v2 * inv; w[r * 4 + 3] = v3 * inv;
}

__global__ void k_combine(const float* __restrict__ h, const float* __restrict__ msgS,
                          const float* __restrict__ msgA, const float* __restrict__ anyv,
                          const float* __restrict__ mi, const float* __restrict__ mr,
                          const float* __restrict__ wb, const float* __restrict__ relsc,
                          const float* __restrict__ inc, const unsigned char* __restrict__ mask,
                          float* __restrict__ xout)
{
  using namespace cfg;
  const int i = blockIdx.x * blockDim.x + threadIdx.x;
  if (i >= BN * S * D) return;
  const int bns = i / D, d = i % D;
  const int b = bns / (NE * S);
  const int rem = bns % (NE * S);
  const int n = rem / S, s = rem % S;
  const int bn = b * NE + n;
  const float* wv = wb + (size_t)bn * 4;
  const float mf = mask[bn] ? 1.f : 0.f;
  const float insc = (inc[(size_t)bn * S + s] * mf > 0.f) ? 1.f : 0.f;
  const float c = wv[0] * msgS[i]
                + wv[1] * msgA[i] * anyv[bns]
                + wv[2] * mi[(size_t)(b * S + s) * D + d] * insc
                + wv[3] * relsc[0] * mr[(size_t)bn * D + d];
  xout[i] = h[(size_t)bn * D + d] + c;
}

// ---------------------------------------------------------------------------
// Mamba layer kernels
// ---------------------------------------------------------------------------
__global__ void k_conv_silu(const float* __restrict__ xz, const float* __restrict__ cw,
                            const float* __restrict__ cb, float* __restrict__ x_act)
{
  using namespace cfg;
  const int i = blockIdx.x * blockDim.x + threadIdx.x;
  if (i >= TOK * DI) return;
  const int tok = i / DI, c = i % DI;
  const int s = tok % S;
  const int base = tok - s;
  float acc = cb[c];
  #pragma unroll
  for (int j = 0; j < DC; ++j) {
    const int sj = s + j - (DC - 1);
    if (sj >= 0) acc += xz[(size_t)(base + sj) * (2 * DI) + c] * cw[c * DC + j];
  }
  x_act[i] = siluf_(acc);
}

__global__ void k_delta(const float* __restrict__ xs, const float* __restrict__ dtw,
                        const float* __restrict__ dtb, const float* __restrict__ esc,
                        const float* __restrict__ emo, float* __restrict__ delta)
{
  using namespace cfg;
  const int i = blockIdx.x * blockDim.x + threadIdx.x;
  if (i >= TOK * DI) return;
  const int tok = i / DI, c = i % DI;
  const float v = xs[(size_t)tok * XP] * dtw[c] + dtb[c];
  const float sp = (v > 20.f) ? v : log1pf(__expf(v));
  delta[i] = fmaxf(sp + esc[0] * emo[tok], 1e-4f);
}

__global__ void k_scan(const float* __restrict__ delta, const float* __restrict__ xs,
                       const float* __restrict__ x_act, float* __restrict__ ybuf)
{
  using namespace cfg;
  const int tid = blockIdx.x * blockDim.x + threadIdx.x;
  if (tid >= BN * DI) return;
  const int bn = tid / DI, c = tid % DI;
  float h[DSS];
  #pragma unroll
  for (int q = 0; q < DSS; ++q) h[q] = 0.f;
  for (int t = 0; t < S; ++t) {
    const int tok = bn * S + t;
    const float e  = __expf(-delta[(size_t)tok * DI + c]);
    const float xv = x_act[(size_t)tok * DI + c];
    const float* xr = xs + (size_t)tok * XP;
    float y = 0.f;
    #pragma unroll
    for (int q = 0; q < DSS; ++q) {
      float hv = h[q] * e + xr[1 + q] * xv;
      hv = fminf(fmaxf(hv, -100.f), 100.f);
      h[q] = hv;
      y += hv * xr[1 + DSS + q];
    }
    ybuf[(size_t)tok * DI + c] = y;
  }
}

__global__ void k_ygate(const float* __restrict__ ybuf, const float* __restrict__ xz,
                        float* __restrict__ yz)
{
  using namespace cfg;
  const int i = blockIdx.x * blockDim.x + threadIdx.x;
  if (i >= TOK * DI) return;
  const int tok = i / DI, c = i % DI;
  const float z = xz[(size_t)tok * (2 * DI) + DI + c];
  yz[i] = ybuf[i] * siluf_(z);
}

// ---------------------------------------------------------------------------
// Final head
// ---------------------------------------------------------------------------
__global__ void k_take_last(const float* __restrict__ x, float* __restrict__ lastb)
{
  using namespace cfg;
  const int i = blockIdx.x * blockDim.x + threadIdx.x;
  if (i >= BN * D) return;
  const int bn = i / D, d = i % D;
  lastb[i] = x[(size_t)(bn * S + (S - 1)) * D + d];
}

__global__ void k_add2(const float* __restrict__ a, const float* __restrict__ b,
                       float* __restrict__ y, int n)
{
  const int i = blockIdx.x * blockDim.x + threadIdx.x;
  if (i >= n) return;
  y[i] = a[i] + b[i];
}

__global__ void k_maskout(const float* __restrict__ x, const unsigned char* __restrict__ mask,
                          float* __restrict__ out)
{
  using namespace cfg;
  const int i = blockIdx.x * blockDim.x + threadIdx.x;
  if (i >= BN * D) return;
  out[i] = x[i] * (mask[i / D] ? 1.f : 0.f);
}

// ---------------------------------------------------------------------------
// Host orchestration
// ---------------------------------------------------------------------------
extern "C" void kernel_launch(void* const* d_in, const int* in_sizes, int n_in,
                              void* d_out, int out_size, void* d_ws, size_t ws_size,
                              hipStream_t stream)
{
  (void)in_sizes; (void)n_in; (void)out_size; (void)ws_size;
  using namespace cfg;

  const float* scene = (const float*)d_in[0];
  const float* inc   = (const float*)d_in[1];
  const int*   edget = (const int*)d_in[2];
  const int*   entt  = (const int*)d_in[3];
  const unsigned char* mask = (const unsigned char*)d_in[4];
  const float* namee = (const float*)d_in[5];
  const float* emo   = (const float*)d_in[6];

  enum {
    P_ENT_TYPE_EMB = 0, P_INIT_W, P_INIT_B, P_INIT_G, P_INIT_BE,
    P_NAME_W, P_NAME_G, P_NAME_BE,
    P_EE_W, P_EE_B, P_ES_W, P_ES_B, P_HE_TYPE_EMB, P_EN_G, P_EN_BE,
    P_SQ_W, P_SQ_B, P_SK_W, P_SK_B, P_SV_W, P_SV_B, P_SM_G, P_SM_BE,
    P_AQ_W, P_AQ_B, P_AK_W, P_AK_B, P_AV_W, P_AV_B, P_AN_G, P_AN_BE,
    P_INT_W, P_INT_B, P_INT_G, P_INT_BE,
    P_REL_W, P_REL_B, P_REL_G, P_REL_BE, P_REL_SCALE,
    P_G1_W, P_G1_B, P_G2_W, P_G2_B,
    P_MAMBA0,                                 // 10 tensors per layer, 2 layers
    P_UPD_G = P_MAMBA0 + 20, P_UPD_BE, P_RES_G, P_RES_BE
  };
  auto P = [&](int i) -> const float* { return (const float*)d_in[7 + i]; };

  char* wsp = (char*)d_ws;
  auto alloc = [&](size_t nf) -> float* {
    float* r = (float*)wsp;
    wsp += ((nf * sizeof(float) + 255) & ~(size_t)255);
    return r;
  };

  auto gemm = [&](const float* X, const float* W, const float* bias, const float* res,
                  float* Y, int M, int Nn, int K) {
    const dim3 grid((Nn + 15) / 16, (M + 63) / 64);
    k_gemm_wmma<<<grid, dim3(128), 0, stream>>>(X, W, bias, res, Y, M, Nn, K);
  };
  auto L1D = [&](int total) { return dim3((total + 255) / 256); };
  const dim3 T256(256);

  // ---- workspace ----
  float* led   = alloc((size_t)BN * D);
  float* t1    = alloc((size_t)BN * D);
  float* namep = alloc((size_t)BN * D);
  float* h     = alloc((size_t)BN * D);
  float* me    = alloc((size_t)BS * D);
  float* tA    = alloc((size_t)BS * D);
  float* tB    = alloc((size_t)BS * D);
  float* eall  = alloc((size_t)BS * D);
  float* qS    = alloc((size_t)BN * D);
  float* kS    = alloc((size_t)BS * D);
  float* vS    = alloc((size_t)BS * D);
  float* attnS = alloc((size_t)BN * S);
  float* arcS  = alloc((size_t)BN * S);
  float* aarc  = alloc((size_t)BN * S * S);
  float* anyv  = alloc((size_t)BN * S);
  float* mi    = alloc((size_t)BS * D);
  float* mr    = alloc((size_t)BN * D);
  float* gb    = alloc((size_t)BN * (D / 4));
  float* wb    = alloc((size_t)BN * 4);
  float* msgS  = alloc((size_t)TOK * D);
  float* msgA  = alloc((size_t)TOK * D);
  float* x     = alloc((size_t)TOK * D);
  float* xz    = alloc((size_t)TOK * 2 * DI);
  float* xact  = alloc((size_t)TOK * DI);
  float* xsb   = alloc((size_t)TOK * XP);
  float* dlt   = alloc((size_t)TOK * DI);
  float* ybuf  = alloc((size_t)TOK * DI);
  float* yzb   = alloc((size_t)TOK * DI);
  float* lastb = alloc((size_t)BN * D);
  float* lnl   = alloc((size_t)BN * D);
  float* sumb  = alloc((size_t)BN * D);

  // ---- h_init ----
  k_led<<<L1D(BN * D), T256, 0, stream>>>(inc, scene, led);
  gemm(led, P(P_INIT_W), P(P_INIT_B), nullptr, t1, BN, D, D);
  k_layernorm<<<BN, T256, 0, stream>>>(t1, P(P_INIT_G), P(P_INIT_BE), t1, BN, D);
  gemm(namee, P(P_NAME_W), nullptr, nullptr, namep, BN, D, D);
  k_layernorm<<<BN, T256, 0, stream>>>(namep, P(P_NAME_G), P(P_NAME_BE), namep, BN, D);
  k_hinit_comb<<<L1D(BN * D), T256, 0, stream>>>(t1, namep, P(P_ENT_TYPE_EMB), entt, mask, h);

  // ---- edge features e_all ----
  k_mean_ent<<<L1D(BS * D), T256, 0, stream>>>(inc, mask, h, me);
  gemm(me, P(P_EE_W), P(P_EE_B), nullptr, tA, BS, D, D);
  gemm(scene, P(P_ES_W), P(P_ES_B), nullptr, tB, BS, D, D);
  k_eall_pre<<<L1D(BS * D), T256, 0, stream>>>(tA, tB, P(P_HE_TYPE_EMB), edget, eall);
  k_layernorm<<<BS, T256, 0, stream>>>(eall, P(P_EN_G), P(P_EN_BE), eall, BS, D);

  const float scale = 1.f / sqrtf((float)D);

  // ---- scene message ----
  gemm(h, P(P_SQ_W), P(P_SQ_B), nullptr, qS, BN, D, D);
  gemm(eall, P(P_SK_W), P(P_SK_B), nullptr, kS, BS, D, D);
  gemm(eall, P(P_SV_W), P(P_SV_B), nullptr, vS, BS, D, D);
  k_dot<<<L1D(B * NE * S), T256, 0, stream>>>(qS, kS, attnS, NE, S, scale);
  k_mpre<<<L1D(TOK * D), T256, 0, stream>>>(attnS, vS, msgS);
  k_layernorm<<<TOK, T256, 0, stream>>>(msgS, P(P_SM_G), P(P_SM_BE), msgS, TOK, D);

  // ---- arc message (reuse q/k/v buffers) ----
  gemm(h, P(P_AQ_W), P(P_AQ_B), nullptr, qS, BN, D, D);
  gemm(eall, P(P_AK_W), P(P_AK_B), nullptr, kS, BS, D, D);
  gemm(eall, P(P_AV_W), P(P_AV_B), nullptr, vS, BS, D, D);
  k_dot<<<L1D(B * NE * S), T256, 0, stream>>>(qS, kS, arcS, NE, S, scale);
  k_arc_softmax<<<L1D(BN * S), T256, 0, stream>>>(arcS, inc, mask, aarc, anyv);
  k_arc_msg<<<L1D(TOK * D), T256, 0, stream>>>(aarc, vS, msgA);
  k_layernorm<<<TOK, T256, 0, stream>>>(msgA, P(P_AN_G), P(P_AN_BE), msgA, TOK, D);

  // ---- interaction & relational messages ----
  gemm(me, P(P_INT_W), P(P_INT_B), nullptr, mi, BS, D, D);
  k_layernorm<<<BS, T256, 0, stream>>>(mi, P(P_INT_G), P(P_INT_BE), mi, BS, D);
  gemm(h, P(P_REL_W), P(P_REL_B), nullptr, mr, BN, D, D);
  k_layernorm<<<BN, T256, 0, stream>>>(mr, P(P_REL_G), P(P_REL_BE), mr, BN, D);

  // ---- gating ----
  gemm(h, P(P_G1_W), P(P_G1_B), nullptr, gb, BN, D / 4, D);
  k_gelu<<<L1D(BN * (D / 4)), T256, 0, stream>>>(gb, BN * (D / 4));
  gemm(gb, P(P_G2_W), P(P_G2_B), nullptr, wb, BN, 4, D / 4);
  k_softmax4<<<L1D(BN), T256, 0, stream>>>(wb, BN);

  // ---- trajectory ----
  k_combine<<<L1D(TOK * D), T256, 0, stream>>>(h, msgS, msgA, anyv, mi, mr, wb,
                                               P(P_REL_SCALE), inc, mask, x);

  // ---- Mamba layers (ping-pong: msgS/msgA buffers reused after combine) ----
  float* cur = x;
  float* nxt = msgS;
  float* xn  = msgA;
  for (int l = 0; l < 2; ++l) {
    const int pb = P_MAMBA0 + l * 10;
    k_layernorm<<<TOK, T256, 0, stream>>>(cur, P(pb + 0), P(pb + 1), xn, TOK, D);
    gemm(xn, P(pb + 2), nullptr, nullptr, xz, TOK, 2 * DI, D);
    k_conv_silu<<<L1D(TOK * DI), T256, 0, stream>>>(xz, P(pb + 3), P(pb + 4), xact);
    gemm(xact, P(pb + 5), nullptr, nullptr, xsb, TOK, XP, DI);
    k_delta<<<L1D(TOK * DI), T256, 0, stream>>>(xsb, P(pb + 6), P(pb + 7), P(pb + 9), emo, dlt);
    k_scan<<<L1D(BN * DI), T256, 0, stream>>>(dlt, xsb, xact, ybuf);
    k_ygate<<<L1D(TOK * DI), T256, 0, stream>>>(ybuf, xz, yzb);
    gemm(yzb, P(pb + 8), nullptr, cur, nxt, TOK, D, DI);
    float* t = cur; cur = nxt; nxt = t;
  }

  // ---- final head ----
  k_take_last<<<L1D(BN * D), T256, 0, stream>>>(cur, lastb);
  k_layernorm<<<BN, T256, 0, stream>>>(lastb, P(P_UPD_G), P(P_UPD_BE), lnl, BN, D);
  k_add2<<<L1D(BN * D), T256, 0, stream>>>(h, lnl, sumb, BN * D);
  k_layernorm<<<BN, T256, 0, stream>>>(sumb, P(P_RES_G), P(P_RES_BE), sumb, BN, D);
  k_maskout<<<L1D(BN * D), T256, 0, stream>>>(sumb, mask, (float*)d_out);
}
